// KVCacheOptimizedHead_61598420959611
// MI455X (gfx1250) — matrix operand order repaired and checked by reading, not compiled, not run on hardware
//
#include <hip/hip_runtime.h>

#define BATCH  4
#define SEQ    8192
#define DMODEL 768
#define HDIM   64

typedef __attribute__((ext_vector_type(16))) __bf16 v16bf;
typedef __attribute__((ext_vector_type(8)))  __bf16 v8bf;
typedef __attribute__((ext_vector_type(4)))  __bf16 v4bf;
typedef __attribute__((ext_vector_type(8)))  float  v8f;

static __device__ __forceinline__ v16bf cat8bf(v8bf lo, v8bf hi) {
    return __builtin_shufflevector(lo, hi, 0,1,2,3,4,5,6,7,8,9,10,11,12,13,14,15);
}
static __device__ __forceinline__ v8f wmma_bf16(v16bf a, v16bf b, v8f c) {
    return __builtin_amdgcn_wmma_f32_16x16x32_bf16(false, a, false, b,
                                                   (short)0, c, false, false);
}

// ---------------------------------------------------------------------------
// Kernel 0: one-time weight convert + swizzle into WMMA B-fragment order.
// Ws[kk][wave][lane][j] = W[kk*32 + (lane>=16?8:0) + (j&7) + (j>=8?16:0)][wave*16 + lane%16]
// so each lane's 16-element fragment is a single contiguous 32B load.
// ---------------------------------------------------------------------------
__global__ __launch_bounds__(256)
void weight_swizzle_kernel(const float* __restrict__ Wq,
                           const float* __restrict__ Wk,
                           const float* __restrict__ Wv,
                           __bf16* __restrict__ Wqs,
                           __bf16* __restrict__ Wks,
                           __bf16* __restrict__ Wvs)
{
    const int t    = blockIdx.x * 256 + threadIdx.x;   // 49152 total
    const int j    = t & 15;
    const int lane = (t >> 4) & 31;
    const int w    = (t >> 9) & 3;
    const int kk   = t >> 11;
    const int row  = kk * 32 + (lane >> 4) * 8 + (j & 7) + (j >> 3) * 16;
    const int col  = w * 16 + (lane & 15);
    const size_t src = (size_t)row * HDIM + col;
    Wqs[t] = (__bf16)Wq[src];
    Wks[t] = (__bf16)Wk[src];
    Wvs[t] = (__bf16)Wv[src];
}

// ---------------------------------------------------------------------------
// Kernel 1: QKV projection.  Block = one 16-row tile of x, 4 waves each own
// one 16-wide column tile.  x staged in LDS as bf16 (padded stride 40 ->
// conflict-free b128 A-fragment gathers).  Weights come pre-swizzled.
// Q,K row-major bf16 [B][S][64]; V TRANSPOSED bf16 [B][64][S].
// ---------------------------------------------------------------------------
__global__ __launch_bounds__(128)
void qkv_proj_kernel(const float*  __restrict__ x,
                     const __bf16* __restrict__ Wqs,
                     const __bf16* __restrict__ Wks,
                     const __bf16* __restrict__ Wvs,
                     __bf16* __restrict__ Qb,
                     __bf16* __restrict__ Kb,
                     __bf16* __restrict__ Vt)
{
    __shared__ __bf16 xs[16][40];                 // padded row stride
    const int tile = blockIdx.x;
    const int b    = tile / (SEQ / 16);
    const int s0   = (tile % (SEQ / 16)) * 16;
    const int tid  = threadIdx.x;
    const int wave = tid >> 5;
    const int lane = tid & 31;
    const int ln   = lane & 15;
    const int hv   = lane >> 4;
    const int kb   = hv ? 8 : 0;

    v8f cq = {}, ck = {}, cv = {};
    const float* xrow = x + ((size_t)b * SEQ + s0) * DMODEL;
    const int col = wave * 16 + ln;

    for (int kk = 0; kk < DMODEL / 32; ++kk) {
        __syncthreads();
        {   // stage + convert x tile [16 x 32] (one float4 per thread)
            const int idx = tid * 4;
            const int r = idx >> 5, c = idx & 31;
            const float4 v = *(const float4*)(xrow + (size_t)r * DMODEL + kk * 32 + c);
            v4bf t;
            t[0] = (__bf16)v.x; t[1] = (__bf16)v.y;
            t[2] = (__bf16)v.z; t[3] = (__bf16)v.w;
            *(v4bf*)&xs[r][c] = t;
        }
        __syncthreads();

        // A fragment: two 16B LDS loads (16-bit A 16x32 layout)
        const v16bf a = cat8bf(*(const v8bf*)&xs[ln][kb],
                               *(const v8bf*)&xs[ln][kb + 16]);

        // B fragments: one contiguous 32B global load each (pre-swizzled)
        const size_t woff = ((size_t)(kk * 4 + wave) * 32 + lane) * 16;
        const v16bf bq = *(const v16bf*)(Wqs + woff);
        const v16bf bk = *(const v16bf*)(Wks + woff);
        const v16bf bv = *(const v16bf*)(Wvs + woff);

        cq = wmma_bf16(a, bq, cq);
        ck = wmma_bf16(a, bk, ck);
        cv = wmma_bf16(a, bv, cv);
    }

    // C layout: element r -> row m = 8*hv + r, col = ln (within n-tile)
    const int mb = hv * 8;
    #pragma unroll
    for (int r = 0; r < 8; ++r) {
        const size_t qoff = ((size_t)b * SEQ + s0 + mb + r) * HDIM + col;
        Qb[qoff] = (__bf16)cq[r];
        Kb[qoff] = (__bf16)ck[r];
    }
    v8bf vpack;
    #pragma unroll
    for (int r = 0; r < 8; ++r) vpack[r] = (__bf16)cv[r];
    // 8 consecutive seq positions -> single 16B store into transposed V
    *(v8bf*)(Vt + ((size_t)b * HDIM + col) * SEQ + s0 + mb) = vpack;
}

// ---------------------------------------------------------------------------
// Kernel 2: causal flash attention, Bc=64.  256 thr = 8 waves; each wave
// owns 16 query rows and streams KV in tiles of 64 with online softmax in
// the exp2 domain (scale*log2e folded into one multiply per element).
// ---------------------------------------------------------------------------
__global__ __launch_bounds__(256)
void flash_attn_kernel(const __bf16* __restrict__ Qb,
                       const __bf16* __restrict__ Kb,
                       const __bf16* __restrict__ Vt,
                       float* __restrict__ out)
{
    __shared__ __bf16 Pl[8][16 * 72];             // per-wave P, padded stride
    const int tid  = threadIdx.x;
    const int wave = tid >> 5;
    const int lane = tid & 31;
    const int ln   = lane & 15;
    const int hv   = lane >> 4;
    const int kb   = hv ? 16 : 0;
    const int kh   = hv ? 16 : 0;
    const int mb   = hv * 8;

    const int blk = blockIdx.x;                   // B * SEQ/128 blocks
    const int b   = blk / (SEQ / 128);
    const int q0  = (blk % (SEQ / 128)) * 128 + wave * 16;

    // Q A-fragments for hd 0..31 and 32..63 (persistent)
    const __bf16* qrow = Qb + ((size_t)b * SEQ + q0 + ln) * HDIM;
    const v16bf qa0 = cat8bf(*(const v8bf*)(qrow + kb),
                             *(const v8bf*)(qrow + kb + 8));
    const v16bf qa1 = cat8bf(*(const v8bf*)(qrow + 32 + kb),
                             *(const v8bf*)(qrow + 32 + kb + 8));

    float mi[8], li[8];
    #pragma unroll
    for (int r = 0; r < 8; ++r) { mi[r] = -3.0e38f; li[r] = 0.0f; }
    v8f o0 = {}, o1 = {}, o2 = {}, o3 = {};

    // scale folded with log2(e): exp(s*scale - m) == exp2(s*scl2 - m2)
    const float scl2 = 0.125f * 1.44269504088896340736f;
    const __bf16* Kbase = Kb + (size_t)b * SEQ * HDIM;
    const __bf16* Vbase = Vt + (size_t)b * HDIM * SEQ;
    __bf16* Pw = Pl[wave];

    const int kend = q0 + 15;
    for (int kv0 = 0; kv0 <= kend; kv0 += 64) {
        // prefetch next KV tile (guarded: only when the next iter exists)
        if (kv0 + 64 <= kend) {
            __builtin_prefetch(Kbase + (size_t)(kv0 + 64 + ln) * HDIM, 0, 1);
            __builtin_prefetch(Vbase + (size_t)ln * SEQ + kv0 + 64, 0, 1);
        }

        // ---- scores: S[16x64] = Q[16x64] @ K^T (8 WMMAs) ----
        v8f s0 = {}, s1 = {}, s2 = {}, s3 = {};
        {
            const __bf16* k0 = Kbase + (size_t)(kv0 +      ln) * HDIM;
            const __bf16* k1 = Kbase + (size_t)(kv0 + 16 + ln) * HDIM;
            const __bf16* k2 = Kbase + (size_t)(kv0 + 32 + ln) * HDIM;
            const __bf16* k3 = Kbase + (size_t)(kv0 + 48 + ln) * HDIM;
            s0 = wmma_bf16(qa0, *(const v16bf*)(k0 + kh), s0);
            s0 = wmma_bf16(qa1, *(const v16bf*)(k0 + 32 + kh), s0);
            s1 = wmma_bf16(qa0, *(const v16bf*)(k1 + kh), s1);
            s1 = wmma_bf16(qa1, *(const v16bf*)(k1 + 32 + kh), s1);
            s2 = wmma_bf16(qa0, *(const v16bf*)(k2 + kh), s2);
            s2 = wmma_bf16(qa1, *(const v16bf*)(k2 + 32 + kh), s2);
            s3 = wmma_bf16(qa0, *(const v16bf*)(k3 + kh), s3);
            s3 = wmma_bf16(qa1, *(const v16bf*)(k3 + 32 + kh), s3);
        }

        // ---- mask + online softmax in exp2 domain (row = mb + r) ----
        float e0[8], e1[8], e2[8], e3[8], corr[8];
        #pragma unroll
        for (int r = 0; r < 8; ++r) {
            const int q = q0 + mb + r;
            float v0 = s0[r] * scl2;
            float v1 = s1[r] * scl2;
            float v2 = s2[r] * scl2;
            float v3 = s3[r] * scl2;
            if (kv0 +      ln > q) v0 = -3.0e38f;
            if (kv0 + 16 + ln > q) v1 = -3.0e38f;
            if (kv0 + 32 + ln > q) v2 = -3.0e38f;
            if (kv0 + 48 + ln > q) v3 = -3.0e38f;
            float t = fmaxf(fmaxf(v0, v1), fmaxf(v2, v3));
            #pragma unroll
            for (int off = 1; off < 16; off <<= 1)
                t = fmaxf(t, __shfl_xor(t, off, 16));
            const float mnew = fmaxf(mi[r], t);
            corr[r] = exp2f(mi[r] - mnew);
            mi[r]   = mnew;
            e0[r] = exp2f(v0 - mnew);
            e1[r] = exp2f(v1 - mnew);
            e2[r] = exp2f(v2 - mnew);
            e3[r] = exp2f(v3 - mnew);
            float rs = (e0[r] + e1[r]) + (e2[r] + e3[r]);
            #pragma unroll
            for (int off = 1; off < 16; off <<= 1)
                rs += __shfl_xor(rs, off, 16);
            li[r] = li[r] * corr[r] + rs;
        }

        // ---- P[16x64]: C-layout -> A-layout via per-wave LDS (bf16) ----
        #pragma unroll
        for (int r = 0; r < 8; ++r) {
            __bf16* prow = Pw + (mb + r) * 72;
            prow[ln]      = (__bf16)e0[r];
            prow[16 + ln] = (__bf16)e1[r];
            prow[32 + ln] = (__bf16)e2[r];
            prow[48 + ln] = (__bf16)e3[r];
        }
        __builtin_amdgcn_wave_barrier();          // keep DS store->load order
        const __bf16* pr = Pw + ln * 72 + kb;
        const v16bf pa0 = cat8bf(*(const v8bf*)pr,
                                 *(const v8bf*)(pr + 8));    // keys 0..31
        const v16bf pa1 = cat8bf(*(const v8bf*)(pr + 32),
                                 *(const v8bf*)(pr + 40));   // keys 32..63

        // ---- O = O*corr + P @ V (8 WMMAs; V frags contiguous 32B) ----
        #pragma unroll
        for (int r = 0; r < 8; ++r) {
            o0[r] *= corr[r]; o1[r] *= corr[r];
            o2[r] *= corr[r]; o3[r] *= corr[r];
        }
        const __bf16* vc0 = Vbase + (size_t)(0  + ln) * SEQ + kv0 + kh;
        const __bf16* vc1 = Vbase + (size_t)(16 + ln) * SEQ + kv0 + kh;
        const __bf16* vc2 = Vbase + (size_t)(32 + ln) * SEQ + kv0 + kh;
        const __bf16* vc3 = Vbase + (size_t)(48 + ln) * SEQ + kv0 + kh;
        o0 = wmma_bf16(pa0, *(const v16bf*)(vc0),      o0);
        o0 = wmma_bf16(pa1, *(const v16bf*)(vc0 + 32), o0);
        o1 = wmma_bf16(pa0, *(const v16bf*)(vc1),      o1);
        o1 = wmma_bf16(pa1, *(const v16bf*)(vc1 + 32), o1);
        o2 = wmma_bf16(pa0, *(const v16bf*)(vc2),      o2);
        o2 = wmma_bf16(pa1, *(const v16bf*)(vc2 + 32), o2);
        o3 = wmma_bf16(pa0, *(const v16bf*)(vc3),      o3);
        o3 = wmma_bf16(pa1, *(const v16bf*)(vc3 + 32), o3);
    }

    // ---- epilogue: normalize by row sum, store fp32 ----
    #pragma unroll
    for (int r = 0; r < 8; ++r) {
        const float inv = 1.0f / li[r];
        float* orow = out + ((size_t)b * SEQ + q0 + mb + r) * HDIM;
        orow[ln]      = o0[r] * inv;
        orow[16 + ln] = o1[r] * inv;
        orow[32 + ln] = o2[r] * inv;
        orow[48 + ln] = o3[r] * inv;
    }
}

// ---------------------------------------------------------------------------
extern "C" void kernel_launch(void* const* d_in, const int* in_sizes, int n_in,
                              void* d_out, int out_size, void* d_ws, size_t ws_size,
                              hipStream_t stream) {
    const float* x  = (const float*)d_in[0];
    const float* Wq = (const float*)d_in[1];
    const float* Wk = (const float*)d_in[2];
    const float* Wv = (const float*)d_in[3];

    const size_t qkv_elems = (size_t)BATCH * SEQ * HDIM;   // 2M elems each
    const size_t w_elems   = (size_t)DMODEL * HDIM;        // 48K elems each

    __bf16* Qb  = (__bf16*)d_ws;
    __bf16* Kb  = Qb + qkv_elems;
    __bf16* Vt  = Kb + qkv_elems;                          // transposed V
    __bf16* Wqs = Vt + qkv_elems;
    __bf16* Wks = Wqs + w_elems;
    __bf16* Wvs = Wks + w_elems;

    weight_swizzle_kernel<<<(int)(w_elems / 256), 256, 0, stream>>>(
        Wq, Wk, Wv, Wqs, Wks, Wvs);
    qkv_proj_kernel<<<BATCH * SEQ / 16, 128, 0, stream>>>(
        x, Wqs, Wks, Wvs, Qb, Kb, Vt);
    flash_attn_kernel<<<BATCH * SEQ / 128, 256, 0, stream>>>(
        Qb, Kb, Vt, (float*)d_out);
}
